// StackedLSTM_3547642987243
// MI455X (gfx1250) — matrix operand order repaired and checked by reading, not compiled, not run on hardware
//
#include <hip/hip_runtime.h>
#include <hip/hip_bf16.h>
#include <stdint.h>

// ---------------------------------------------------------------------------
// Stacked 2-layer LSTM for MI455X (gfx1250), bf16 WMMA with fp32 accumulate.
// Persistent kernel, LDS-resident weights, grid-wide one-shot barriers.
// ---------------------------------------------------------------------------

typedef __attribute__((ext_vector_type(16))) __bf16 v16bf;
typedef __attribute__((ext_vector_type(8)))  float  v8f;
typedef __attribute__((ext_vector_type(4)))  unsigned int u32x4;   // POD 16-byte load

constexpr int Bsz = 32, Tsz = 256, Esz = 1024, Hsz = 1024;
constexpr int NG  = 4 * Hsz;            // 4096 gate columns per layer
constexpr int K2  = Esz + Hsz;          // 2048 concat reduction dim
constexpr int KP  = K2 + 8;             // padded LDS row stride (bf16 elems)
constexpr int GEMM_BLOCKS      = 128;
constexpr int UNITS_PER_BLOCK  = 4;     // (layer, ntile) units cached per block
constexpr size_t SMEM_BYTES = (size_t)UNITS_PER_BLOCK * 16 * KP * sizeof(__bf16); // 263168

__device__ __forceinline__ float sigf(float x)      { return 1.0f / (1.0f + __expf(-x)); }
__device__ __forceinline__ float tanh_fast(float x) { return 1.0f - 2.0f / (__expf(2.0f * x) + 1.0f); }

// One-shot grid barrier (slot `idx` used exactly once per launch; init kernel zeroes).
__device__ __forceinline__ void grid_sync(unsigned* bar, int idx, unsigned nb) {
    __threadfence();              // publish this thread's global writes
    __syncthreads();
    if (threadIdx.x == 0) {
        __hip_atomic_fetch_add(&bar[idx], 1u, __ATOMIC_ACQ_REL, __HIP_MEMORY_SCOPE_AGENT);
        while (__hip_atomic_load(&bar[idx], __ATOMIC_ACQUIRE, __HIP_MEMORY_SCOPE_AGENT) < nb)
            __builtin_amdgcn_s_sleep(2);
    }
    __syncthreads();
}

// ------------------------- prep kernels ------------------------------------

// Zero h/c/hb states and barrier slots; copy mask to output tail.
__global__ void lstm_init(float* stateF, unsigned* hb_u32, unsigned* bar,
                          const float* __restrict__ mask, float* out_mask) {
    int i = blockIdx.x * blockDim.x + threadIdx.x;
    if (i < 4 * Bsz * Hsz) stateF[i] = 0.0f;
    if (i < Bsz * Hsz)     hb_u32[i] = 0u;       // 2*Bsz*Hsz bf16 == Bsz*Hsz dwords
    if (i < 2 * Tsz)       bar[i] = 0u;
    if (i < Bsz * Tsz)     out_mask[i] = mask[i];
}

// Build Wt[layer][n (4096)][k (2048)] bf16, k = [Wih rows ; Whh rows] transposed.
// LDS-tiled 32x32 transpose so both global read and write are coalesced.
__global__ __launch_bounds__(256) void wt_transpose(
        const float* __restrict__ Wih0, const float* __restrict__ Whh0,
        const float* __restrict__ Wih1, const float* __restrict__ Whh1,
        __bf16* __restrict__ Wt) {
    __shared__ float tile[32][33];
    const int bid   = blockIdx.x;          // 4 srcs * 32 ktiles * 128 ntiles
    const int srcId = bid >> 12;
    const int rem   = bid & 4095;
    const int ktile = rem >> 7;            // 0..31
    const int ntile = rem & 127;           // 0..127
    const float* src = (srcId == 0) ? Wih0 : (srcId == 1) ? Whh0
                      : (srcId == 2) ? Wih1 : Whh1;
    const int layer = srcId >> 1;
    const int kbase = (srcId & 1) * Esz;
    const int x = threadIdx.x & 31, y = threadIdx.x >> 5;   // (32,8)
#pragma unroll
    for (int r = 0; r < 4; ++r) {
        int kk = y + r * 8;
        tile[kk][x] = src[(size_t)(ktile * 32 + kk) * NG + ntile * 32 + x];
    }
    __syncthreads();
#pragma unroll
    for (int r = 0; r < 4; ++r) {
        int nn = y + r * 8;
        int n  = ntile * 32 + nn;
        Wt[((size_t)layer * NG + n) * K2 + kbase + ktile * 32 + x] = (__bf16)tile[x][nn];
    }
}

__global__ void x_to_bf16(const float* __restrict__ x, __bf16* __restrict__ xb, int n) {
    for (int i = blockIdx.x * blockDim.x + threadIdx.x; i < n; i += gridDim.x * blockDim.x)
        xb[i] = (__bf16)x[i];
}

// ------------------------- persistent LSTM kernel --------------------------

__global__ __launch_bounds__(256, 1) void lstm_persist(
        const __bf16* __restrict__ xb, const __bf16* __restrict__ Wt,
        const float* __restrict__ b0p, const float* __restrict__ b1p,
        const float* __restrict__ maskp,
        float* __restrict__ stateF, __bf16* __restrict__ hbB,
        float* __restrict__ pre, float* __restrict__ out,
        unsigned* __restrict__ bar) {
    extern __shared__ __align__(16) char smem_raw[];
    __bf16* lds_w = (__bf16*)smem_raw;

    const int tid = threadIdx.x;

    // ---- one-time LDS fill: this block's 4 weight slices (16 cols x 2048 k each)
    for (int i = tid; i < UNITS_PER_BLOCK * 16 * (K2 / 8); i += 256) {
        int row = i >> 8;               // 256 u32x4 per row
        int q   = i & 255;
        int ul  = row >> 4, nr = row & 15;
        int un  = blockIdx.x * UNITS_PER_BLOCK + ul;
        int ly  = un >> 8, nt = un & 255;
        const u32x4* s = (const u32x4*)(Wt + ((size_t)ly * NG + nt * 16 + nr) * K2);
        u32x4* d = (u32x4*)(lds_w + (size_t)(ul * 16 + nr) * KP);
        d[q] = s[q];
    }
    __syncthreads();

    const int wave    = tid >> 5;
    const int lane    = tid & 31;
    const int u_local = wave >> 1;
    const int mt      = wave & 1;
    const int unit    = blockIdx.x * UNITS_PER_BLOCK + u_local;
    const int layer   = unit >> 8;
    const int nt      = unit & 255;
    const int n0      = nt * 16;
    const int laneN   = lane & 15;
    const int hiH     = lane >> 4;

    float* h0 = stateF;
    float* c0 = stateF + Bsz * Hsz;
    float* h1 = stateF + 2 * Bsz * Hsz;
    float* c1 = stateF + 3 * Bsz * Hsz;
    __bf16* hb0 = hbB;
    __bf16* hb1 = hbB + Bsz * Hsz;

    const int brow = mt * 16 + laneN;                       // batch row of A fragment
    const int aoff = hiH * 8;                               // A k-offset per ISA layout
    const __bf16* lwrow = lds_w + (size_t)(u_local * 16 + laneN) * KP + hiH * 8;
    const float bias = (layer ? b1p : b0p)[n0 + laneN];
    float* pbase = pre + (size_t)layer * Bsz * NG
                       + (size_t)(mt * 16 + hiH * 8) * NG + n0 + laneN;

    const int gt = blockIdx.x * 256 + tid;                  // elementwise thread id

    for (int t = 0; t < Tsz; ++t) {
        // ---------------- GEMM phase: pre = [inp ; h_prev] @ Wt + b ----------------
        const __bf16* srcX = layer ? (hb0 + brow * Hsz)
                                   : (xb + ((size_t)brow * Tsz + t) * Esz);
        const __bf16* srcY = layer ? (hb1 + brow * Hsz) : (hb0 + brow * Hsz);

        if (layer == 0 && lane == 0 && t + 1 < Tsz)
            __builtin_prefetch(srcX + Esz, 0, 0);           // next x_t tile -> cache

        // bias folded into accumulator init (added exactly once per output elem)
        v8f acc = {bias, bias, bias, bias, bias, bias, bias, bias};
#pragma unroll 8
        for (int kc = 0; kc < 32; ++kc) {                   // k in [0, 1024): inp half
            const __bf16* sa = srcX + kc * 32 + aoff;
            union { u32x4 u[2]; v16bf v; } A, Bv;
            A.u[0]  = *(const u32x4*)sa;
            A.u[1]  = *(const u32x4*)(sa + 16);
            const __bf16* sb = lwrow + kc * 32;
            Bv.u[0] = *(const u32x4*)sb;
            Bv.u[1] = *(const u32x4*)(sb + 16);
            acc = __builtin_amdgcn_wmma_f32_16x16x32_bf16(false, A.v, false, Bv.v,
                                                          (short)0, acc, false, false);
        }
#pragma unroll 8
        for (int kc = 0; kc < 32; ++kc) {                   // k in [1024, 2048): h half
            const __bf16* sa = srcY + kc * 32 + aoff;
            union { u32x4 u[2]; v16bf v; } A, Bv;
            A.u[0]  = *(const u32x4*)sa;
            A.u[1]  = *(const u32x4*)(sa + 16);
            const __bf16* sb = lwrow + Esz + kc * 32;
            Bv.u[0] = *(const u32x4*)sb;
            Bv.u[1] = *(const u32x4*)(sb + 16);
            acc = __builtin_amdgcn_wmma_f32_16x16x32_bf16(false, A.v, false, Bv.v,
                                                          (short)0, acc, false, false);
        }
#pragma unroll
        for (int r = 0; r < 8; ++r) pbase[(size_t)r * NG] = acc[r];

        grid_sync(bar, 2 * t, GEMM_BLOCKS);

        // ---------------- elementwise phase: gates, state update ------------------
        for (int e = gt; e < 2 * Bsz * Hsz; e += GEMM_BLOCKS * 256) {
            int ly = e >> 15;
            int r  = e & 32767;
            int b  = r >> 10;
            int j  = r & 1023;
            const float* pl = pre + (size_t)ly * Bsz * NG + (size_t)b * NG;
            float f  = pl[j];
            float ig = pl[Hsz + j];
            float og = pl[2 * Hsz + j];
            float gg = pl[3 * Hsz + j];
            float* C  = ly ? c1 : c0;
            float* Hm = ly ? h1 : h0;
            __bf16* HB = ly ? hb1 : hb0;
            int si = b * Hsz + j;
            float c_old = C[si], h_old = Hm[si];
            float mv = maskp[b * Tsz + t];
            float cn = sigf(f) * c_old + sigf(ig) * tanh_fast(gg);
            cn = cn * mv + c_old * (1.0f - mv);
            float hn = sigf(og) * tanh_fast(cn);
            hn = hn * mv + h_old * (1.0f - mv);
            C[si] = cn;
            Hm[si] = hn;
            HB[si] = (__bf16)hn;
            if (ly) out[((size_t)b * Tsz + t) * Hsz + j] = hn;   // states output = h1
        }

        if (t + 1 < Tsz) grid_sync(bar, 2 * t + 1, GEMM_BLOCKS);
    }
}

// ------------------------- host launcher -----------------------------------

extern "C" void kernel_launch(void* const* d_in, const int* in_sizes, int n_in,
                              void* d_out, int out_size, void* d_ws, size_t ws_size,
                              hipStream_t stream) {
    const float* x    = (const float*)d_in[0];
    const float* mask = (const float*)d_in[1];
    const float* Wih0 = (const float*)d_in[2];
    const float* Whh0 = (const float*)d_in[3];
    const float* b0   = (const float*)d_in[4];
    const float* Wih1 = (const float*)d_in[5];
    const float* Whh1 = (const float*)d_in[6];
    const float* b1   = (const float*)d_in[7];
    float* out = (float*)d_out;
    (void)in_sizes; (void)n_in; (void)out_size; (void)ws_size;

    char* ws = (char*)d_ws;
    size_t o = 0;
    auto wsalloc = [&](size_t bytes) -> void* {
        void* p = ws + o;
        o = (o + bytes + 255) & ~(size_t)255;
        return p;
    };
    __bf16*  Wt  = (__bf16*) wsalloc((size_t)2 * NG * K2 * sizeof(__bf16));     // 32 MB
    __bf16*  xb  = (__bf16*) wsalloc((size_t)Bsz * Tsz * Esz * sizeof(__bf16)); // 16 MB
    float*   pre = (float*)  wsalloc((size_t)2 * Bsz * NG * sizeof(float));     // 1 MB
    float*   st  = (float*)  wsalloc((size_t)4 * Bsz * Hsz * sizeof(float));    // h0|c0|h1|c1
    __bf16*  hb  = (__bf16*) wsalloc((size_t)2 * Bsz * Hsz * sizeof(__bf16));   // hb0|hb1
    unsigned* bar = (unsigned*)wsalloc((size_t)2 * Tsz * sizeof(unsigned));

    hipFuncSetAttribute((const void*)lstm_persist,
                        hipFuncAttributeMaxDynamicSharedMemorySize, (int)SMEM_BYTES);

    lstm_init<<<512, 256, 0, stream>>>(st, (unsigned*)hb, bar, mask,
                                       out + (size_t)Bsz * Tsz * Hsz);
    wt_transpose<<<4 * 32 * 128, 256, 0, stream>>>(Wih0, Whh0, Wih1, Whh1, Wt);
    x_to_bf16<<<4096, 256, 0, stream>>>(x, xb, Bsz * Tsz * Esz);
    lstm_persist<<<GEMM_BLOCKS, 256, SMEM_BYTES, stream>>>(
        xb, Wt, b0, b1, mask, st, hb, pre, out, bar);
}
